// LFADS_38628935860789
// MI455X (gfx1250) — matrix operand order repaired and checked by reading, not compiled, not run on hardware
//
#include <hip/hip_runtime.h>
#include <cstddef>

#define TT 512
#define BB 256
#define DD 256
#define HH 256
#define LL 64
#define FF 64

typedef __attribute__((ext_vector_type(16))) __bf16         v16bf;
typedef __attribute__((ext_vector_type(8)))  float          v8f;
typedef __attribute__((ext_vector_type(4)))  unsigned int   u32x4;
typedef __attribute__((ext_vector_type(4)))  float          f32x4;
typedef __attribute__((ext_vector_type(4)))  unsigned short u16x4;

union AFrag { v16bf v; u32x4 q[2]; unsigned int u[8]; };
union BFrag { v16bf v; u32x4 q[2]; };

__device__ __forceinline__ unsigned short f2bf(float f) {
  unsigned int u = __float_as_uint(f);
  u += 0x7FFFu + ((u >> 16) & 1u);        // round-to-nearest-even
  return (unsigned short)(u >> 16);
}

__device__ __forceinline__ float sigmoidf_(float x) { return 1.0f / (1.0f + __expf(-x)); }

// C[16 x NTILES*16] (f32, LDS, stride ldc) = A[16 x K] (bf16 LDS, stride lda) @ W^T
// W is [NTILES*16 x K] bf16 row-major in global memory (W rows = B columns).
// Workgroup = 8 waves; wave w handles output tiles nt = w, w+8, ...
// A fragments are loaded into registers ONCE per wave (they are nt-invariant),
// and the tile loop is kept serial so only one tile's B fragments are live.
template <int K, int NTILES>
__device__ __forceinline__ void wg_gemm(const unsigned short* __restrict__ A, int lda,
                                        const unsigned short* __restrict__ W,
                                        float* __restrict__ C, int ldc) {
  const int lane = threadIdx.x & 31;
  const int wid  = threadIdx.x >> 5;
  const int mcol = lane & 15;   // A-row index for A-frag; tile-column for B/C frags
  const int hx   = lane >> 4;   // half-wave select
  constexpr int KT = K / 32;

  // 16-bit A 16x32 layout (ISA 7.12.2): lane holds row mcol; halves 0..7 are
  // k = kb + hx*8 + 0..7 (contiguous 16B), halves 8..15 are k+16 (contiguous 16B).
  AFrag a[KT];
#pragma unroll
  for (int kt = 0; kt < KT; ++kt) {
    const unsigned short* ap = A + mcol * lda + kt * 32 + hx * 8;
    a[kt].q[0] = *(const u32x4*)(ap);        // k .. k+7
    a[kt].q[1] = *(const u32x4*)(ap + 16);   // k+16 .. k+23
  }

  constexpr int ITER = (NTILES + 7) / 8;
#pragma unroll 1
  for (int i = 0; i < ITER; ++i) {
    const int nt = wid + i * 8;
    if ((NTILES % 8 == 0) || (nt < NTILES)) {
      // 16-bit B 32x16 layout: lane holds column nt*16+mcol, k = kb + hx*8 + 0..7 and kb + 16 + hx*8 + 0..7
      const unsigned short* wrow = W + (size_t)(nt * 16 + mcol) * K + hx * 8;
      v8f acc = {0.f, 0.f, 0.f, 0.f, 0.f, 0.f, 0.f, 0.f};
#pragma unroll
      for (int kt = 0; kt < KT; ++kt) {
        BFrag b;
        const u32x4* wq = (const u32x4*)(wrow + kt * 32);
        b.q[0] = wq[0];
        b.q[1] = wq[2];
        acc = __builtin_amdgcn_wmma_f32_16x16x32_bf16(false, a[kt].v, false, b.v,
                                                      (short)0, acc, false, false);
      }
      // f32 C/D layout: lane = column, VGPR r -> row hx*8 + r
      float* cb = C + nt * 16 + mcol;
#pragma unroll
      for (int r = 0; r < 8; ++r) cb[(hx * 8 + r) * ldc] = acc[r];
    }
  }
}

// PyTorch GRU gate combine: r,z,n order; gi = x@Wih^T, gh = h@Whh^T (separate!)
__device__ __forceinline__ void gru_combine(const float* __restrict__ gi,
                                            const float* __restrict__ gh,
                                            const float* __restrict__ bih,
                                            const float* __restrict__ bhh,
                                            float* __restrict__ h_f,
                                            unsigned short* __restrict__ h_bf, int tid) {
  for (int idx = tid; idx < 16 * 256; idx += 256) {
    const int m = idx >> 8, j = idx & 255;
    const float* gim = gi + m * 768;
    const float* ghm = gh + m * 768;
    const float r = sigmoidf_(gim[j] + bih[j] + ghm[j] + bhh[j]);
    const float z = sigmoidf_(gim[256 + j] + bih[256 + j] + ghm[256 + j] + bhh[256 + j]);
    const float n = tanhf(gim[512 + j] + bih[512 + j] + r * (ghm[512 + j] + bhh[512 + j]));
    const float hv = (1.0f - z) * n + z * h_f[idx];
    h_f[idx]  = hv;
    h_bf[idx] = f2bf(hv);
  }
}

// Stage a 16x256 f32 tile (row stride rowstride in the source) into bf16 LDS
// with 128-bit global loads and 64-bit LDS stores. ldd = LDS row stride.
__device__ __forceinline__ void stage_x_tile(const float* __restrict__ src, size_t rowstride,
                                             unsigned short* __restrict__ dst, int ldd, int tid) {
  for (int idx = tid * 4; idx < 16 * 256; idx += 256 * 4) {
    const int m = idx >> 8, d = idx & 255;
    const f32x4 v = *(const f32x4*)(src + (size_t)m * rowstride + d);
    u16x4 o;
    o.x = f2bf(v.x); o.y = f2bf(v.y); o.z = f2bf(v.z); o.w = f2bf(v.w);
    *(u16x4*)(dst + m * ldd + d) = o;
  }
}

__global__ void f32_to_bf16_kernel(const float* __restrict__ src,
                                   unsigned short* __restrict__ dst, int n) {
  const int i = blockIdx.x * 256 + threadIdx.x;
  if (i < n) dst[i] = f2bf(src[i]);
}

// ---------------- Bidirectional encoder: grid (B/16, 2 dirs), 256 threads ----------------
__global__ __launch_bounds__(256, 1) void lfads_encoder_kernel(
    const float* __restrict__ x,
    const unsigned short* __restrict__ Wih_f, const unsigned short* __restrict__ Whh_f,
    const float* __restrict__ bih_f, const float* __restrict__ bhh_f,
    const unsigned short* __restrict__ Wih_b, const unsigned short* __restrict__ Whh_b,
    const float* __restrict__ bih_b, const float* __restrict__ bhh_b,
    float* __restrict__ h_enc) {
  extern __shared__ char smem[];
  unsigned short* x_bf = (unsigned short*)smem;   // 16x256 bf16
  unsigned short* h_bf = x_bf + 16 * 256;         // 16x256 bf16
  float* h_f = (float*)(h_bf + 16 * 256);         // 16x256 f32 (master state)
  float* gi  = h_f + 16 * 256;                    // 16x768 f32
  float* gh  = gi + 16 * 768;                     // 16x768 f32   (total 128 KB)

  const int tid = threadIdx.x;
  const int b0  = blockIdx.x * 16;
  const int dir = blockIdx.y;
  const unsigned short* Wih = dir ? Wih_b : Wih_f;
  const unsigned short* Whh = dir ? Whh_b : Whh_f;
  const float* bih = dir ? bih_b : bih_f;
  const float* bhh = dir ? bhh_b : bhh_f;

  for (int idx = tid; idx < 16 * 256; idx += 256) { h_f[idx] = 0.f; h_bf[idx] = 0; }
  __syncthreads();

  for (int t = 0; t < TT; ++t) {
    const int te = dir ? (TT - 1 - t) : t;
    stage_x_tile(x + ((size_t)b0 * TT + te) * DD, (size_t)TT * DD, x_bf, 256, tid);
    __syncthreads();
    wg_gemm<256, 48>(x_bf, 256, Wih, gi, 768);
    wg_gemm<256, 48>(h_bf, 256, Whh, gh, 768);
    __syncthreads();
    gru_combine(gi, gh, bih, bhh, h_f, h_bf, tid);
    __syncthreads();
  }
  for (int idx = tid; idx < 16 * 256; idx += 256) {
    const int m = idx >> 8, j = idx & 255;
    h_enc[(size_t)(b0 + m) * 512 + dir * 256 + j] = h_f[idx];
  }
}

// ---------------- IC heads + kl_ics block partials (tiny) ----------------
__global__ __launch_bounds__(64) void lfads_ic_kernel(
    const float* __restrict__ h_enc, const float* __restrict__ Wmu,
    const float* __restrict__ bmu, const float* __restrict__ Wlv,
    const float* __restrict__ blv, float* __restrict__ icp) {
  const int b = blockIdx.x, l = threadIdx.x;
  const float* h  = h_enc + (size_t)b * 512;
  const float* wm = Wmu + (size_t)l * 512;
  const float* wl = Wlv + (size_t)l * 512;
  float mu = bmu[l], lv = blv[l];
  for (int k = 0; k < 512; ++k) { const float hv = h[k]; mu += hv * wm[k]; lv += hv * wl[k]; }
  __shared__ float red[64];
  red[l] = 0.5f * (__expf(lv) + mu * mu - 1.0f - lv);
  __syncthreads();
  for (int s = 32; s > 0; s >>= 1) { if (l < s) red[l] += red[l + s]; __syncthreads(); }
  if (l == 0) icp[b] = red[0];
}

// ---------------- Controller + generator recurrence: 16 persistent WGs ----------------
__global__ __launch_bounds__(256, 1) void lfads_ctrlgen_kernel(
    const float* __restrict__ x, const float* __restrict__ eps_u,
    const unsigned short* __restrict__ Wc_ih, const unsigned short* __restrict__ Wc_hh,
    const float* __restrict__ bc_ih, const float* __restrict__ bc_hh,
    const unsigned short* __restrict__ Wcmu, const float* __restrict__ bcmu,
    const unsigned short* __restrict__ Wclv, const float* __restrict__ bclv,
    const unsigned short* __restrict__ Wg_ih, const unsigned short* __restrict__ Wg_hh,
    const float* __restrict__ bg_ih, const float* __restrict__ bg_hh,
    const unsigned short* __restrict__ Wf, const float* __restrict__ bf_,
    const unsigned short* __restrict__ Wr, const float* __restrict__ br_,
    float* __restrict__ rates, float* __restrict__ factors, float* __restrict__ ctp) {
  extern __shared__ char smem[];
  unsigned short* a_bf  = (unsigned short*)smem;  // 16x320 controller input (x_t | prev_f)
  unsigned short* ch_bf = a_bf + 16 * 320;        // 16x256
  unsigned short* g_bf  = ch_bf + 16 * 256;       // 16x256
  unsigned short* u_bf  = g_bf + 16 * 256;        // 16x64
  unsigned short* f_bf  = u_bf + 16 * 64;         // 16x64
  float* ch_f = (float*)(f_bf + 16 * 64);         // 16x256
  float* g_f  = ch_f + 16 * 256;                  // 16x256
  float* fac  = g_f + 16 * 256;                   // 16x64  (prev_f, persists across t)
  float* gi   = fac + 16 * 64;                    // 16x768 scratch
  float* gh   = gi + 16 * 768;                    // 16x768 scratch  (~162 KB total)
  float* cmu    = gi;             // overlays (gi free between GRU stages)
  float* clv    = gi + 16 * 64;
  float* facraw = gi;
  float* rr     = gh;
  __shared__ float red[256];

  const int tid = threadIdx.x;
  const int b0  = blockIdx.x * 16;

  for (int idx = tid; idx < 16 * 256; idx += 256) {
    ch_f[idx] = 0.f; ch_bf[idx] = 0; g_f[idx] = 0.f; g_bf[idx] = 0;
  }
  for (int idx = tid; idx < 16 * 64; idx += 256) fac[idx] = 0.f;
  __syncthreads();

  float kl = 0.f;
  for (int t = 0; t < TT; ++t) {
    // build c_in = [x_t | prev_f]
    stage_x_tile(x + ((size_t)b0 * TT + t) * DD, (size_t)TT * DD, a_bf, 320, tid);
    for (int idx = tid; idx < 16 * 64; idx += 256) {
      const int m = idx >> 6, j = idx & 63;
      a_bf[m * 320 + 256 + j] = f2bf(fac[idx]);
    }
    __syncthreads();
    // controller GRU
    wg_gemm<320, 48>(a_bf, 320, Wc_ih, gi, 768);
    wg_gemm<256, 48>(ch_bf, 256, Wc_hh, gh, 768);
    __syncthreads();
    gru_combine(gi, gh, bc_ih, bc_hh, ch_f, ch_bf, tid);
    __syncthreads();
    // posterior heads
    wg_gemm<256, 4>(ch_bf, 256, Wcmu, cmu, 64);
    wg_gemm<256, 4>(ch_bf, 256, Wclv, clv, 64);
    __syncthreads();
    // reparameterized sample + KL accumulation
    for (int idx = tid; idx < 16 * 64; idx += 256) {
      const int m = idx >> 6, l = idx & 63;
      const float mu = cmu[idx] + bcmu[l];
      const float lv = clv[idx] + bclv[l];
      const float ep = eps_u[((size_t)t * BB + (b0 + m)) * LL + l];
      u_bf[idx] = f2bf(mu + ep * __expf(0.5f * lv));
      kl += 0.5f * (__expf(lv) + mu * mu - 1.0f - lv);
    }
    __syncthreads();
    // generator GRU
    wg_gemm<64, 48>(u_bf, 64, Wg_ih, gi, 768);
    wg_gemm<256, 48>(g_bf, 256, Wg_hh, gh, 768);
    __syncthreads();
    gru_combine(gi, gh, bg_ih, bg_hh, g_f, g_bf, tid);
    __syncthreads();
    // factors readout
    wg_gemm<256, 4>(g_bf, 256, Wf, facraw, 64);
    __syncthreads();
    for (int idx = tid; idx < 16 * 64; idx += 256) {
      const int m = idx >> 6, l = idx & 63;
      const float fv = facraw[idx] + bf_[l];
      fac[idx]  = fv;
      f_bf[idx] = f2bf(fv);
      factors[((size_t)(b0 + m) * TT + t) * FF + l] = fv;
    }
    __syncthreads();
    // rates = exp(factors @ W_r^T + b_r)
    wg_gemm<64, 16>(f_bf, 64, Wr, rr, 256);
    __syncthreads();
    for (int idx = tid; idx < 16 * 256; idx += 256) {
      const int m = idx >> 8, d = idx & 255;
      rates[((size_t)(b0 + m) * TT + t) * DD + d] = __expf(rr[idx] + br_[d]);
    }
    __syncthreads();
  }
  red[tid] = kl;
  __syncthreads();
  for (int s = 128; s > 0; s >>= 1) { if (tid < s) red[tid] += red[tid + s]; __syncthreads(); }
  if (tid == 0) ctp[blockIdx.x] = red[0];
}

__global__ void lfads_finalize_kernel(const float* __restrict__ icp,
                                      const float* __restrict__ ctp,
                                      float* __restrict__ kli, float* __restrict__ klc) {
  if (threadIdx.x == 0 && blockIdx.x == 0) {
    float s = 0.f;
    for (int i = 0; i < 256; ++i) s += icp[i];
    *kli = s / 256.0f;                       // mean over B
    float c = 0.f;
    for (int i = 0; i < 16; ++i) c += ctp[i];
    *klc = c / (256.0f * 512.0f);            // mean over B and T
  }
}

extern "C" void kernel_launch(void* const* d_in, const int* in_sizes, int n_in,
                              void* d_out, int out_size, void* d_ws, size_t ws_size,
                              hipStream_t stream) {
  (void)in_sizes; (void)n_in; (void)out_size; (void)ws_size;
  const float* x      = (const float*)d_in[0];
  const float* eps_u  = (const float*)d_in[1];
  const float* Wih_ef = (const float*)d_in[2];
  const float* Whh_ef = (const float*)d_in[3];
  const float* bih_ef = (const float*)d_in[4];
  const float* bhh_ef = (const float*)d_in[5];
  const float* Wih_eb = (const float*)d_in[6];
  const float* Whh_eb = (const float*)d_in[7];
  const float* bih_eb = (const float*)d_in[8];
  const float* bhh_eb = (const float*)d_in[9];
  const float* W_mu0  = (const float*)d_in[10];
  const float* b_mu0  = (const float*)d_in[11];
  const float* W_lv0  = (const float*)d_in[12];
  const float* b_lv0  = (const float*)d_in[13];
  const float* Wih_c  = (const float*)d_in[14];
  const float* Whh_c  = (const float*)d_in[15];
  const float* bih_c  = (const float*)d_in[16];
  const float* bhh_c  = (const float*)d_in[17];
  const float* W_cmu  = (const float*)d_in[18];
  const float* b_cmu  = (const float*)d_in[19];
  const float* W_clv  = (const float*)d_in[20];
  const float* b_clv  = (const float*)d_in[21];
  const float* Wih_g  = (const float*)d_in[22];
  const float* Whh_g  = (const float*)d_in[23];
  const float* bih_g  = (const float*)d_in[24];
  const float* bhh_g  = (const float*)d_in[25];
  const float* W_f    = (const float*)d_in[26];
  const float* b_f    = (const float*)d_in[27];
  const float* W_r    = (const float*)d_in[28];
  const float* b_r    = (const float*)d_in[29];

  // ---- workspace layout: bf16 weights (128B-aligned) then f32 buffers ----
  unsigned short* wbf = (unsigned short*)d_ws;
  size_t off = 0;
  auto alloc_bf = [&](int n) {
    unsigned short* p = wbf + off;
    off += (size_t)((n + 63) & ~63);
    return p;
  };
  unsigned short* bWih_ef = alloc_bf(768 * 256);
  unsigned short* bWhh_ef = alloc_bf(768 * 256);
  unsigned short* bWih_eb = alloc_bf(768 * 256);
  unsigned short* bWhh_eb = alloc_bf(768 * 256);
  unsigned short* bWih_c  = alloc_bf(768 * 320);
  unsigned short* bWhh_c  = alloc_bf(768 * 256);
  unsigned short* bWcmu   = alloc_bf(64 * 256);
  unsigned short* bWclv   = alloc_bf(64 * 256);
  unsigned short* bWih_g  = alloc_bf(768 * 64);
  unsigned short* bWhh_g  = alloc_bf(768 * 256);
  unsigned short* bWf     = alloc_bf(64 * 256);
  unsigned short* bWr     = alloc_bf(256 * 64);
  float* h_enc = (float*)(wbf + off);      // [B][512]
  float* icp   = h_enc + 256 * 512;        // [256] kl_ics partials
  float* ctp   = icp + 256;                // [16]  kl_ctrl partials

  auto cvt = [&](const float* s, unsigned short* d, int n) {
    f32_to_bf16_kernel<<<(n + 255) / 256, 256, 0, stream>>>(s, d, n);
  };
  cvt(Wih_ef, bWih_ef, 768 * 256);
  cvt(Whh_ef, bWhh_ef, 768 * 256);
  cvt(Wih_eb, bWih_eb, 768 * 256);
  cvt(Whh_eb, bWhh_eb, 768 * 256);
  cvt(Wih_c,  bWih_c,  768 * 320);
  cvt(Whh_c,  bWhh_c,  768 * 256);
  cvt(W_cmu,  bWcmu,   64 * 256);
  cvt(W_clv,  bWclv,   64 * 256);
  cvt(Wih_g,  bWih_g,  768 * 64);
  cvt(Whh_g,  bWhh_g,  768 * 256);
  cvt(W_f,    bWf,     64 * 256);
  cvt(W_r,    bWr,     256 * 64);

  lfads_encoder_kernel<<<dim3(16, 2), 256, 131072, stream>>>(
      x, bWih_ef, bWhh_ef, bih_ef, bhh_ef, bWih_eb, bWhh_eb, bih_eb, bhh_eb, h_enc);

  lfads_ic_kernel<<<256, 64, 0, stream>>>(h_enc, W_mu0, b_mu0, W_lv0, b_lv0, icp);

  float* out     = (float*)d_out;
  float* rates   = out;
  float* kli     = out + (size_t)BB * TT * DD;
  float* klc     = kli + 1;
  float* factors = klc + 1;

  lfads_ctrlgen_kernel<<<16, 256, 165888, stream>>>(
      x, eps_u, bWih_c, bWhh_c, bih_c, bhh_c, bWcmu, b_cmu, bWclv, b_clv,
      bWih_g, bWhh_g, bih_g, bhh_g, bWf, b_f, bWr, b_r, rates, factors, ctp);

  lfads_finalize_kernel<<<1, 1, 0, stream>>>(icp, ctp, kli, klc);
}
